// LinearPredictiveCoding_11759620456913
// MI455X (gfx1250) — matrix operand order repaired and checked by reading, not compiled, not run on hardware
//
#include <hip/hip_runtime.h>
#include <hip/hip_bf16.h>

typedef __attribute__((ext_vector_type(16))) _Float16 v16h;
typedef __attribute__((ext_vector_type(8)))  float    v8f;

#define NUM_CODE 32
#define WIN      1024
#define STRIDE   256
#define BATCH    32
#define TLEN     524288
#define NFRAMES  2049            // TLEN/STRIDE + 1
#define FPB      8               // frames per block (one per wave)
#define NGROUPS  257             // ceil(2049/8)
#define SEGLEN   2816            // (FPB-1)*STRIDE + WIN
#define EPS      1e-7f

// corr[k] = sum_n x[n] * x[(n+k) mod 1024]  (circular autocorrelation)
// Gram trick: C[m][n] = sum_t x[(t+m)%1024] * x[(t+2n+2)%1024] = corr[2n+2-m]
// d = 2n+2-m covers all of 0..32 within a single 16x16 WMMA tile.

union BFrag {                    // B dword-pair view: u[e2] = {h[2e2] lo, h[2e2+1] hi}
    v16h     h;
    unsigned u[8];
};

__global__ __launch_bounds__(256)
void lpc_wmma_kernel(const float* __restrict__ in, float* __restrict__ out) {
    __shared__ _Float16 seg[SEGLEN];        // shared f16 copy of covering segment
    __shared__ float    ctile[FPB][256];    // per-wave 16x16 f32 WMMA result

    const int tid   = threadIdx.x;
    const int lane  = tid & 31;
    const int w     = tid >> 5;             // wave id = frame-in-group
    const int g     = blockIdx.x;           // frame group
    const int b     = blockIdx.y;           // batch

    // ---- stage covering segment (f32 global -> f16 LDS), zero-pad tail ----
    const long segStart = (long)g * (FPB * STRIDE);
    const long bBase    = (long)b * TLEN;
    for (int i = tid; i < SEGLEN; i += 256) {
        long pos = segStart + i;
        float v = (pos < TLEN) ? in[bBase + pos] : 0.0f;
        seg[i] = (_Float16)v;
    }
    __syncthreads();

    // ---- per-wave frame (unclamped; all waves compute, store guarded by fTrue) ----
    const int fTrue = g * FPB + w;
    const int wbase = w * STRIDE;           // frame start inside segment (even)

    // A fragment map (this code's contiguous k order, 16-bit A 16x32):
    //   lane<16: M=lane, K=e ; lanes>=16: +16 on K
    // B fragment map (dense 32x16, per sparse-B half layout):
    //   lanes 0-15: N=lane, K=e ; lanes 16-31: N=lane-16, K=e+16
    const int m    = lane & 15;             // A row shift (mixed parity -> u16 gathers)
    const int kloA = (lane & 16) ? 16 : 0;
    const int khiB = (lane & 16) ? 16 : 0;
    const int bsh  = 2 * (lane & 15) + 2;   // B col shift = 2n+2 (even -> b32 gathers)

    const unsigned* segU = reinterpret_cast<const unsigned*>(seg);

    v8f acc0, acc1;
#pragma unroll
    for (int i = 0; i < 8; ++i) { acc0[i] = 0.0f; acc1[i] = 0.0f; }

    // K = 1024 in 32 chunks of 32; two independent WMMA chains to hide the
    // D->C RAW hazard (5 NOPs for f16 per ISA 7.12.1).
    for (int c = 0; c < 32; c += 2) {
#pragma unroll
        for (int half = 0; half < 2; ++half) {
            const int t0 = (c + half) << 5;
            v16h a;
            BFrag bf;
#pragma unroll
            for (int e = 0; e < 16; ++e) {
                const int ka = e + kloA;
                a[e] = seg[wbase + ((t0 + ka + m) & (WIN - 1))];
            }
#pragma unroll
            for (int e2 = 0; e2 < 8; ++e2) {
                // idx is even: t0 even, 2*e2+khiB even, bsh even -> 4B aligned
                const int idx = (t0 + 2 * e2 + khiB + bsh) & (WIN - 1);
                bf.u[e2] = segU[(wbase + idx) >> 1];
            }
            if (half == 0)
                acc0 = __builtin_amdgcn_wmma_f32_16x16x32_f16(
                           false, a, false, bf.h, (short)0, acc0, false, false);
            else
                acc1 = __builtin_amdgcn_wmma_f32_16x16x32_f16(
                           false, a, false, bf.h, (short)0, acc1, false, false);
        }
    }

    // ---- spill C tile: lane<16 -> M=r, lane>=16 -> M=8+r ; N=lane&15 ----
    {
        const int mhi = (lane >> 4) << 3;
        const int nn  = lane & 15;
#pragma unroll
        for (int r = 0; r < 8; ++r)
            ctile[w][(r + mhi) * 16 + nn] = acc0[r] + acc1[r];
    }
    __syncthreads();

    // ---- extract corr[d] for d = lane (0..31), plus corr[32] scalar ----
    // canonical (m,n) with 2n+2-m = d:  d<=2 -> (2-d, 0) ; d>=3 -> n=(d-1)>>1
    int em, en;
    if (lane <= 2) { em = 2 - lane; en = 0; }
    else           { en = (lane - 1) >> 1; em = 2 * en + 2 - lane; }
    float corrReg = ctile[w][em * 16 + en]; // lane i holds corr[i]
    float c32     = ctile[w][0 * 16 + 15];  // d=32 -> (m=0, n=15)

    // ---- wave-parallel Levinson-Durbin (reference semantics) ----
    const float corr0 = __shfl(corrReg, 0);
    const float corr1 = __shfl(corrReg, 1);
    const float c0    = fmaxf(corr0, EPS);
    const float sol1  = -corr1 / c0;
    float sol   = (lane == 0) ? 1.0f : ((lane == 1) ? sol1 : 0.0f);
    float extra = corr0 + corr1 * sol1;
    float lamLast = 0.0f;

    for (int k = 1; k < NUM_CODE; ++k) {
        const int idx = k + 1 - lane;       // corr index for this lane's sol term
        // gather corr[k+1-lane]; shuffles executed unconditionally (EXEC full)
        float sh = __shfl(corrReg, ((unsigned)idx) & 31);
        float cv = (idx == 32) ? c32 : sh;
        float contrib = (lane <= k) ? sol * cv : 0.0f;
        float s = contrib;
#pragma unroll
        for (int off = 16; off > 0; off >>= 1) s += __shfl_xor(s, off);
        const float lam = -s / fmaxf(extra, EPS);
        // sol_new[i] = aug[i] + lam*aug[k+1-i]   (aug = sol padded with 0)
        float flipRaw = __shfl(sol, ((unsigned)idx) & 31);
        float flip = (idx == 32) ? 0.0f : flipRaw;   // aug[32] == 0
        sol = (lane <= k + 1) ? (sol + lam * flip) : 0.0f;
        extra = (1.0f - lam * lam) * extra;
        lamLast = lam;
    }
    // sol[32] = lamLast * sol[0] = lamLast  (sol[0] stays 1 throughout)

    // ---- output sol[1..32]: lane j writes sol[j+1] ----
    float ov = __shfl(sol, (lane + 1) & 31);
    if (lane == 31) ov = lamLast;
    if (fTrue < NFRAMES) {
        long o = ((long)b * NFRAMES + fTrue) * NUM_CODE + lane;
        out[o] = ov;
    }
}

extern "C" void kernel_launch(void* const* d_in, const int* in_sizes, int n_in,
                              void* d_out, int out_size, void* d_ws, size_t ws_size,
                              hipStream_t stream) {
    const float* in = (const float*)d_in[0];
    float* out = (float*)d_out;
    dim3 grid(NGROUPS, BATCH, 1);
    dim3 block(256, 1, 1);
    lpc_wmma_kernel<<<grid, block, 0, stream>>>(in, out);
}
